// SelfAlignLayer_11862699671625
// MI455X (gfx1250) — matrix operand order repaired and checked by reading, not compiled, not run on hardware
//
#include <hip/hip_runtime.h>

typedef __attribute__((ext_vector_type(16))) __bf16 v16bf;
typedef __attribute__((ext_vector_type(8)))  __bf16 v8bf;
typedef __attribute__((ext_vector_type(8)))  float  v8f;
typedef __attribute__((ext_vector_type(4)))  unsigned int u32x4;
typedef __attribute__((ext_vector_type(8)))  unsigned int u32x8;

#define B_ 16
#define L_ 1024
#define D_ 768

// ---------------------------------------------------------------------------
// Kernel 1: P = relu(emb @ W^T + b), fp32 in -> bf16 out (WMMA f32 accum)
// One wave per 16x16 tile, K = 768 (24 bf16 WMMAs).
// ---------------------------------------------------------------------------
__global__ __launch_bounds__(32)
void proj_relu_kernel(const float* __restrict__ emb, const float* __restrict__ W,
                      const float* __restrict__ bias, __bf16* __restrict__ P) {
  const int lane = threadIdx.x & 31;
  const int m    = lane & 15;
  const int half = lane >> 4;
  const int row0 = blockIdx.x * 16;   // over B*L
  const int col0 = blockIdx.y * 16;   // over D

  const float* arow = emb + (size_t)(row0 + m) * D_;
  const float* brow = W   + (size_t)(col0 + m) * D_;   // B[k][n] = W[n][k]

  v8f c = {};
  for (int k0 = 0; k0 < D_; k0 += 32) {
    v16bf a, b;
#pragma unroll
    for (int e = 0; e < 8; ++e) {
      a[e]     = (__bf16)arow[k0 + half * 8 + e];
      a[8 + e] = (__bf16)arow[k0 + 16 + half * 8 + e];
    }
#pragma unroll
    for (int e = 0; e < 16; ++e) b[e] = (__bf16)brow[k0 + (e & 8) * 2 + half * 8 + (e & 7)];
    c = __builtin_amdgcn_wmma_f32_16x16x32_bf16(false, a, false, b, (short)0, c,
                                                false, false);
  }

  const int   col = col0 + m;
  const float bv  = bias[col];
#pragma unroll
  for (int v = 0; v < 8; ++v) {
    float r = c[v] + bv;
    r = r > 0.f ? r : 0.f;
    P[(size_t)(row0 + v + half * 8) * D_ + col] = (__bf16)r;
  }
}

// ---------------------------------------------------------------------------
// Kernel 2: per (batch b, 16-query-row stripe):
//   TDM-stage Prow -> LDS (padded), scores = Prow @ P^T (diag/mask) -> alphaOut
//   softmax over keys -> alphaOut + bf16 LDS copy
//   align = alpha @ emb with LDS-transposed V chunks
// 128 threads = 4 waves.
// ---------------------------------------------------------------------------
#define PROW_BYTES   (16 * D_ * 2)                 // 24576 data bytes
#define PROW_PADDED  (PROW_BYTES + (PROW_BYTES / 1024) * 16)  // +16B per KiB
#define ALB_W        1032                          // 1032*2B row = 516 dwords (bank stride 4)
#define VT_W         40                            // 40*2B row = 20 dwords (bank stride 20)

__global__ __launch_bounds__(128)
void attn_kernel(const __bf16* __restrict__ P, const float* __restrict__ emb,
                 const unsigned char* __restrict__ mask,
                 float* __restrict__ alignOut, float* __restrict__ alphaOut) {
  const int b       = blockIdx.x >> 6;
  const int rt      = blockIdx.x & 63;
  const int rowbase = rt * 16;
  const int tid     = threadIdx.x;
  const int wave    = tid >> 5;
  const int lane    = tid & 31;
  const int m       = lane & 15;
  const int half    = lane >> 4;

  __shared__ alignas(16) unsigned char ProwRaw[PROW_PADDED];  // ~24.4 KB
  __shared__ alignas(16) __bf16 alphaB[16][ALB_W];            // ~33 KB
  __shared__ alignas(16) __bf16 Vt[D_][VT_W];                 // 60 KB (transposed V chunk)

  // ---- Tensor Data Mover: DMA the 16 query rows of P into LDS ----
  // 2D tensor, data_size = 2B, tile = 12288 x 1 elements, contiguous in global.
  // pad_enable: insert 4 dwords (16B) after every 256 dwords (1024B) in LDS.
  if (wave == 0) {
    const unsigned long long ga =
        (unsigned long long)(const void*)(P + ((size_t)b * L_ + rowbase) * D_);
    const unsigned int lds = (unsigned int)(size_t)(void*)&ProwRaw[0];
    u32x4 g0;
    g0[0] = 1u;                                        // count=1, user mode
    g0[1] = lds;                                       // lds_addr
    g0[2] = (unsigned int)(ga & 0xffffffffu);          // global_addr[31:0]
    g0[3] = (unsigned int)((ga >> 32) & 0x1ffffffu) | (2u << 30);  // addr hi | type=2
    u32x8 g1;
    g1[0] = (1u << 16)            // data_size = 2B
          | (1u << 20)            // pad_enable
          | (7u << 22)            // pad_interval = 256 dwords
          | (3u << 25);           // pad_amount = 4 dwords
    g1[1] = (12288u & 0xffffu) << 16;   // tensor_dim0[15:0] = 12288
    g1[2] = (12288u >> 16) | (1u << 16);// tensor_dim0[31:16] | tensor_dim1 = 1
    g1[3] = (12288u << 16);             // tile_dim0 = 12288
    g1[4] = 1u;                         // tile_dim1 = 1, tile_dim2 = 0
    g1[5] = 12288u;                     // tensor_dim0_stride[31:0]
    g1[6] = 0u;
    g1[7] = 0u;
    asm volatile("tensor_load_to_lds %0, %1" ::"s"(g0), "s"(g1) : "memory");
    __builtin_amdgcn_s_wait_tensorcnt(0);
  }
  __syncthreads();

  float* alphaRowBase = alphaOut + ((size_t)b * L_ + rowbase) * L_;

  // ---- scores: S = Prow * Pcol^T over K = 768, raw masked -> alphaOut ----
  for (int ct = wave; ct < L_ / 16; ct += 4) {
    const __bf16* pcol = P + ((size_t)b * L_ + ct * 16 + m) * D_;
    if (ct + 4 < L_ / 16)
      __builtin_prefetch(P + ((size_t)b * L_ + (ct + 4) * 16 + m) * D_, 0, 3);
    v8f c = {};
    for (int k0 = 0; k0 < D_; k0 += 32) {
      v16bf a, bb;
      {
        // padded LDS offsets: off + 16 * (off >> 10); pads never split a 16B chunk
        const int d1 = m * (D_ * 2) + k0 * 2 + half * 16;
        const int d2 = d1 + 32;
        v8bf lo = *(const v8bf*)(ProwRaw + d1 + ((d1 >> 10) << 4));
        v8bf hi = *(const v8bf*)(ProwRaw + d2 + ((d2 >> 10) << 4));
#pragma unroll
        for (int i = 0; i < 8; ++i) { a[i] = lo[i]; a[8 + i] = hi[i]; }
      }
      {
        const __bf16* bp = pcol + k0 + half * 8;    // two 16B chunks, k and k+16
        v8bf b0 = *(const v8bf*)bp;
        v8bf b1 = *(const v8bf*)(bp + 16);
#pragma unroll
        for (int i = 0; i < 8; ++i) { bb[i] = b0[i]; bb[8 + i] = b1[i]; }
      }
      c = __builtin_amdgcn_wmma_f32_16x16x32_bf16(false, a, false, bb, (short)0, c,
                                                  false, false);
    }
    const int  col = ct * 16 + m;
    const bool mk  = mask[(size_t)b * L_ + col] != 0;
#pragma unroll
    for (int v = 0; v < 8; ++v) {
      const int l = rowbase + v + half * 8;
      float s = c[v];
      if (col == l) s = 0.f;
      if (mk) s = -__builtin_inff();
      alphaRowBase[(size_t)(v + half * 8) * L_ + col] = s;
    }
  }
  __threadfence();
  __syncthreads();

  // ---- softmax over keys: 8 lanes per row ----
  {
    const int r   = tid >> 3;
    const int sub = tid & 7;
    float* srow = alphaRowBase + (size_t)r * L_;
    float mx = -__builtin_inff();
    for (int cI = sub; cI < L_; cI += 8) mx = fmaxf(mx, srow[cI]);
#pragma unroll
    for (int o = 1; o < 8; o <<= 1) mx = fmaxf(mx, __shfl_xor(mx, o, 32));
    float sm = 0.f;
    for (int cI = sub; cI < L_; cI += 8) sm += __expf(srow[cI] - mx);
#pragma unroll
    for (int o = 1; o < 8; o <<= 1) sm += __shfl_xor(sm, o, 32);
    const float inv = 1.f / sm;
    for (int cI = sub; cI < L_; cI += 8) {
      const float av = __expf(srow[cI] - mx) * inv;
      srow[cI]      = av;
      alphaB[r][cI] = (__bf16)av;
    }
  }

  // ---- align = alpha @ emb: k-chunk outer, V staged transposed in LDS ----
  v8f acc[12];
#pragma unroll
  for (int j = 0; j < 12; ++j) acc[j] = (v8f){};

  for (int k0 = 0; k0 < L_; k0 += 32) {
    __syncthreads();   // protect Vt against previous iteration's readers
    // stage emb rows [k0, k0+32) of batch b, transposed: Vt[col][k] = emb[k][col]
    {
      const float* src = emb + ((size_t)b * L_ + k0) * D_;
      for (int i = tid; i < 32 * (D_ / 4); i += 128) {
        const int kr   = i / (D_ / 4);
        const int cb4  = i % (D_ / 4);
        const float4 f = ((const float4*)(src + (size_t)kr * D_))[cb4];
        const int col  = cb4 * 4;
        Vt[col + 0][kr] = (__bf16)f.x;
        Vt[col + 1][kr] = (__bf16)f.y;
        Vt[col + 2][kr] = (__bf16)f.z;
        Vt[col + 3][kr] = (__bf16)f.w;
      }
    }
    __syncthreads();

    // A fragment (alpha) — one load per wave per k-step, reused by 12 tiles
    v16bf a;
    {
      const __bf16* ap = &alphaB[m][k0];
      v8bf lo = *(const v8bf*)(ap + half * 8);
      v8bf hi = *(const v8bf*)(ap + 16 + half * 8);
#pragma unroll
      for (int i = 0; i < 8; ++i) { a[i] = lo[i]; a[8 + i] = hi[i]; }
    }

#pragma unroll
    for (int j = 0; j < 12; ++j) {
      const int col = (wave * 12 + j) * 16 + m;
      v16bf bb;
      const __bf16* vp = &Vt[col][half * 8];    // two 16B chunks, k and k+16
      v8bf b0 = *(const v8bf*)vp;
      v8bf b1 = *(const v8bf*)(vp + 16);
#pragma unroll
      for (int i = 0; i < 8; ++i) { bb[i] = b0[i]; bb[8 + i] = b1[i]; }
      acc[j] = __builtin_amdgcn_wmma_f32_16x16x32_bf16(false, a, false, bb, (short)0,
                                                       acc[j], false, false);
    }
  }

#pragma unroll
  for (int j = 0; j < 12; ++j) {
    const int col = (wave * 12 + j) * 16 + m;
#pragma unroll
    for (int v = 0; v < 8; ++v)
      alignOut[((size_t)b * L_ + rowbase + v + half * 8) * D_ + col] = acc[j][v];
  }
}

// ---------------------------------------------------------------------------
extern "C" void kernel_launch(void* const* d_in, const int* in_sizes, int n_in,
                              void* d_out, int out_size, void* d_ws, size_t ws_size,
                              hipStream_t stream) {
  const float*         emb  = (const float*)d_in[0];
  const unsigned char* mask = (const unsigned char*)d_in[1];  // jax bool_ = 1 byte
  const float*         W    = (const float*)d_in[2];
  const float*         bias = (const float*)d_in[3];

  float* alignOut = (float*)d_out;                   // [B, L, D]
  float* alphaOut = alignOut + (size_t)B_ * L_ * D_; // [B, L, L]
  __bf16* P = (__bf16*)d_ws;                         // 24 MB bf16 scratch

  proj_relu_kernel<<<dim3(B_ * L_ / 16, D_ / 16), 32, 0, stream>>>(emb, W, bias, P);
  attn_kernel<<<dim3(B_ * L_ / 16), 128, 0, stream>>>(P, emb, mask, alignOut, alphaOut);
}
